// SelfAttention_67104569033420
// MI455X (gfx1250) — matrix operand order repaired and checked by reading, not compiled, not run on hardware
//
#include <hip/hip_runtime.h>

// Problem constants (match reference)
#define B_   4
#define T_   4096
#define NIN  256
#define DD   128      // n_out
#define WAVES 4
#define QROWS (WAVES * 16)   // 64 query rows per block
#define KT   64              // keys per K/V tile

typedef __attribute__((ext_vector_type(16))) __bf16       v16bf;
typedef __attribute__((ext_vector_type(8)))  float        v8f;
typedef __attribute__((ext_vector_type(4)))  unsigned int v4u;
typedef __attribute__((ext_vector_type(8)))  unsigned int v8u;
typedef __attribute__((ext_vector_type(4)))  float        v4f;
typedef __attribute__((ext_vector_type(8)))  int          tdm8i;
typedef __attribute__((ext_vector_type(4)))  int          tdm4i;

// fp32 -> bf16 round-to-nearest-even (bit trick)
static __device__ __forceinline__ unsigned short f2bf(float x) {
    unsigned int u = __builtin_bit_cast(unsigned int, x);
    u += 0x7FFFu + ((u >> 16) & 1u);
    return (unsigned short)(u >> 16);
}

// Build a 16-element bf16 fragment from two 16-byte chunks (p0: 8 bf16, p1: 8 bf16)
static __device__ __forceinline__ v16bf frag_from(const unsigned short* p0,
                                                  const unsigned short* p1) {
    union { v8u u; v16bf b; } f;
    v4u a = *(const v4u*)p0;
    v4u c = *(const v4u*)p1;
    f.u[0] = a[0]; f.u[1] = a[1]; f.u[2] = a[2]; f.u[3] = a[3];
    f.u[4] = c[0]; f.u[5] = c[1]; f.u[6] = c[2]; f.u[7] = c[3];
    return f.b;
}

// ---------------------------------------------------------------------------
// TDM: DMA a 64-row x 128-col bf16 tile (row stride 128 elems) into LDS.
// D# bitfields per CDNA5 ISA. 2-D tensor: groups
// 2/3 are zero. data_size=1 (2 bytes); type=2 ("image") in group0[127:126].
// This toolchain exposes the 6-arg builtin: (g0, g1, g2, g3, g4, cpol).
// ---------------------------------------------------------------------------
static __device__ __forceinline__ void tdm_load_tile(unsigned lds_off,
                                                     const unsigned short* gptr) {
    const unsigned long long ga = (unsigned long long)(uintptr_t)gptr;
    v4u g0;
    g0[0] = 1u;                                    // count=1, user mode
    g0[1] = lds_off;                               // LDS byte address
    g0[2] = (unsigned)(ga & 0xFFFFFFFFull);        // global_addr[31:0]
    g0[3] = (unsigned)(ga >> 32) | (2u << 30);     // global_addr[56:32] | type=2
    tdm8i g1;
    g1[0] = (int)(1u << 16);     // wg_mask=0, data_size=1 (2 bytes)
    g1[1] = (int)(128u << 16);   // tensor_dim0[15:0]=128 in bits 63:48
    g1[2] = (int)(64u << 16);    // tensor_dim0 hi=0; tensor_dim1[15:0]=64 (bits 95:80)
    g1[3] = (int)(128u << 16);   // tensor_dim1 hi=0; tile_dim0=128 (bits 127:112)
    g1[4] = 64;                  // tile_dim1=64 (bits 143:128); tile_dim2=0
    g1[5] = 128;                 // tensor_dim0_stride[31:0]=128
    g1[6] = 0;                   // stride hi, tensor_dim1_stride lo
    g1[7] = 0;
    const tdm4i z4 = {0, 0, 0, 0};
    const tdm8i z8 = {0, 0, 0, 0, 0, 0, 0, 0};
    __builtin_amdgcn_tensor_load_to_lds(g0, g1, z4, z4, z8, 0);
}

// ---------------------------------------------------------------------------
// Prep 1: seq fp32 -> bf16 (vectorized 4 elems/thread)
// ---------------------------------------------------------------------------
__global__ __launch_bounds__(256)
void cvt_seq_kernel(const float* __restrict__ seq, unsigned short* __restrict__ seqb) {
    const size_t q = (size_t)blockIdx.x * 256 + threadIdx.x;   // quad index
    const v4f x = *(const v4f*)(seq + q * 4);
    unsigned short* o = seqb + q * 4;
    o[0] = f2bf(x[0]); o[1] = f2bf(x[1]); o[2] = f2bf(x[2]); o[3] = f2bf(x[3]);
}

// ---------------------------------------------------------------------------
// Prep 2: W[kd][n] fp32 -> Wt[n][kd] bf16 for the 3 weight matrices.
// ---------------------------------------------------------------------------
__global__ __launch_bounds__(256)
void wt_transpose_kernel(const float* __restrict__ Wq,
                         const float* __restrict__ Wk,
                         const float* __restrict__ Wv,
                         unsigned short* __restrict__ Wt) {
    const int idx = blockIdx.x * 256 + threadIdx.x;   // 0 .. 3*NIN*DD-1
    const int mat = idx / (NIN * DD);
    const int rem = idx % (NIN * DD);
    const int n   = rem / NIN;
    const int kd  = rem % NIN;
    const float* W = (mat == 0) ? Wq : ((mat == 1) ? Wk : Wv);
    Wt[(size_t)mat * NIN * DD + (size_t)n * NIN + kd] = f2bf(W[(size_t)kd * DD + n]);
}

// ---------------------------------------------------------------------------
// Kernel 3: QKV projection on WMMA. 4 waves/block, wave = 16 tokens.
// ---------------------------------------------------------------------------
__global__ __launch_bounds__(128)
void qkv_wmma_kernel(const unsigned short* __restrict__ seqb,
                     const unsigned short* __restrict__ Wt,
                     unsigned short* __restrict__ qb,
                     unsigned short* __restrict__ kb,
                     unsigned short* __restrict__ vb) {
    const int tid  = threadIdx.x;
    const int wave = tid >> 5;
    const int lane = tid & 31;
    const int m    = lane & 15;
    const int h    = lane >> 4;

    const int tb = blockIdx.x * QROWS + wave * 16;   // global token base (bt index)

    // A fragments: 8 x (16 tokens x 32 kd)
    const unsigned short* srow = seqb + (size_t)(tb + m) * NIN;
    v16bf aqs[8];
#pragma unroll
    for (int kt = 0; kt < 8; ++kt) {
        const unsigned short* p0 = srow + kt * 32 + h * 8;
        aqs[kt] = frag_from(p0, p0 + 16);
    }

    const v8f vzero = {0.f, 0.f, 0.f, 0.f, 0.f, 0.f, 0.f, 0.f};
    unsigned short* outs[3] = {qb, kb, vb};

#pragma unroll
    for (int mat = 0; mat < 3; ++mat) {
        const unsigned short* wtm = Wt + (size_t)mat * NIN * DD;
        unsigned short* om = outs[mat];
#pragma unroll
        for (int f = 0; f < 8; ++f) {
            const unsigned short* wrow = wtm + (size_t)(f * 16 + m) * NIN;
            v16bf bw[8];
#pragma unroll
            for (int kt = 0; kt < 8; ++kt) {
                const unsigned short* p = wrow + kt * 32 + h * 8;
                bw[kt] = frag_from(p, p + 16);
            }
            v8f acc = vzero;
#pragma unroll
            for (int kt = 0; kt < 8; ++kt)
                acc = __builtin_amdgcn_wmma_f32_16x16x32_bf16(
                          false, aqs[kt], false, bw[kt], (short)0, acc, false, false);
#pragma unroll
            for (int r = 0; r < 8; ++r)
                om[(size_t)(tb + r + 8 * h) * DD + f * 16 + m] = f2bf(acc[r]);
        }
    }
}

// ---------------------------------------------------------------------------
// Kernel 4: flash attention with TDM double-buffered K/V tiles.
// Wave 0 issues tensor_load_to_lds for tile j+1 while all waves compute on
// tile j. V arrives row-major and is transposed LDS->LDS for B-fragments.
// ---------------------------------------------------------------------------
__global__ __launch_bounds__(WAVES * 32)
void attn_kernel(const unsigned short* __restrict__ qb,
                 const unsigned short* __restrict__ kb,
                 const unsigned short* __restrict__ vb,
                 float* __restrict__ out) {
    __shared__ unsigned short lds_k[2][KT][DD];      // K tiles, row-major  (32 KB)
    __shared__ unsigned short lds_vraw[2][KT][DD];   // V tiles, row-major  (32 KB)
    __shared__ unsigned short lds_v[DD][KT];         // V transposed [e][k] (16 KB)
    __shared__ unsigned short lds_p[WAVES][16][KT];  // per-wave P staging  ( 8 KB)

    const int tid  = threadIdx.x;
    const int wave = tid >> 5;
    const int lane = tid & 31;
    const int m    = lane & 15;   // A-matrix row / B,C column within half
    const int h    = lane >> 4;   // lane half

    const int blk   = blockIdx.x;            // 0 .. B*T/QROWS-1
    const int b     = blk / (T_ / QROWS);
    const int qtile = blk % (T_ / QROWS);
    const int qbase = qtile * QROWS + wave * 16;

    // Q A-fragments (4 x 16x32 over e), held in VGPRs for whole kernel
    const unsigned short* qrowp = qb + (size_t)(b * T_ + qbase + m) * DD;
    v16bf aq[4];
#pragma unroll
    for (int et = 0; et < 4; ++et) {
        const unsigned short* p0 = qrowp + et * 32 + h * 8;
        aq[et] = frag_from(p0, p0 + 16);
    }

    const v8f vzero = {0.f, 0.f, 0.f, 0.f, 0.f, 0.f, 0.f, 0.f};
    v8f o[8];
#pragma unroll
    for (int f = 0; f < 8; ++f) o[f] = vzero;
    float mrow[8], lrow[8];
#pragma unroll
    for (int r = 0; r < 8; ++r) { mrow[r] = -3.0e38f; lrow[r] = 0.f; }

    const unsigned short* kbat = kb + (size_t)b * T_ * DD;
    const unsigned short* vbat = vb + (size_t)b * T_ * DD;

    const unsigned k_off[2] = {(unsigned)(uintptr_t)&lds_k[0][0][0],
                               (unsigned)(uintptr_t)&lds_k[1][0][0]};
    const unsigned v_off[2] = {(unsigned)(uintptr_t)&lds_vraw[0][0][0],
                               (unsigned)(uintptr_t)&lds_vraw[1][0][0]};

    // prologue: DMA tile 0 into buffer 0
    if (wave == 0) {
        tdm_load_tile(k_off[0], kbat);
        tdm_load_tile(v_off[0], vbat);
    }

    for (int j = 0; j < T_ / KT; ++j) {
        const int buf = j & 1;
        if (wave == 0) __builtin_amdgcn_s_wait_tensorcnt(0);
        __syncthreads();                          // tile j resident in LDS

        // prefetch tile j+1 into the other buffer (overlaps all compute below)
        if (wave == 0 && j + 1 < T_ / KT) {
            tdm_load_tile(k_off[buf ^ 1], kbat + (size_t)((j + 1) * KT) * DD);
            tdm_load_tile(v_off[buf ^ 1], vbat + (size_t)((j + 1) * KT) * DD);
        }

        // --- transpose V tile LDS->LDS: lds_v[e][k] = vraw[k][e] ---
        for (int ch = tid; ch < KT * DD / 8; ch += WAVES * 32) {
            const int row = ch >> 4;            // key index
            const int col = (ch & 15) * 8;      // e base
            v4u d = *(const v4u*)&lds_vraw[buf][row][col];
            const unsigned short* ds = (const unsigned short*)&d;
#pragma unroll
            for (int e2 = 0; e2 < 8; ++e2)
                lds_v[col + e2][row] = ds[e2];
        }

        // --- S = Q K^T : 4 key sub-tiles x 4 e sub-tiles of WMMA ---
        v8f s[4];
#pragma unroll
        for (int kt = 0; kt < 4; ++kt) {
            v16bf bk[4];
#pragma unroll
            for (int et = 0; et < 4; ++et) {
                const unsigned short* p0 = &lds_k[buf][kt * 16 + m][et * 32 + h * 8];
                bk[et] = frag_from(p0, p0 + 16);
            }
            v8f acc = vzero;
#pragma unroll
            for (int et = 0; et < 4; ++et)
                acc = __builtin_amdgcn_wmma_f32_16x16x32_bf16(
                          false, aq[et], false, bk[et], (short)0, acc, false, false);
            s[kt] = acc;
        }

        __syncthreads();                          // lds_v fully transposed

        // --- online softmax (rows live in C layout: row = r + 8*h) ---
        float pr[4][8];
#pragma unroll
        for (int r = 0; r < 8; ++r) {
            float mx = fmaxf(fmaxf(s[0][r], s[1][r]), fmaxf(s[2][r], s[3][r]));
            mx = fmaxf(mx, __shfl_xor(mx, 1));
            mx = fmaxf(mx, __shfl_xor(mx, 2));
            mx = fmaxf(mx, __shfl_xor(mx, 4));
            mx = fmaxf(mx, __shfl_xor(mx, 8));
            const float mnew = fmaxf(mrow[r], mx);
            const float scl  = __expf(mrow[r] - mnew);
            mrow[r] = mnew;
            float psum = 0.f;
#pragma unroll
            for (int kt = 0; kt < 4; ++kt) {
                const float p = __expf(s[kt][r] - mnew);
                pr[kt][r] = p;
                psum += p;
            }
            psum += __shfl_xor(psum, 1);
            psum += __shfl_xor(psum, 2);
            psum += __shfl_xor(psum, 4);
            psum += __shfl_xor(psum, 8);
            lrow[r] = lrow[r] * scl + psum;
#pragma unroll
            for (int f = 0; f < 8; ++f) o[f][r] *= scl;   // rescale O accum
        }

        // --- re-layout P (C layout -> A layout) via wave-private LDS ---
#pragma unroll
        for (int kt = 0; kt < 4; ++kt)
#pragma unroll
            for (int r = 0; r < 8; ++r)
                lds_p[wave][r + 8 * h][kt * 16 + m] = f2bf(pr[kt][r]);
        asm volatile("s_wait_dscnt 0" ::: "memory");      // same-wave DS RAW guard

        // --- O += P V : 2 key sub-tiles (K=32) x 8 e tiles ---
        v16bf ap[2];
#pragma unroll
        for (int kt2 = 0; kt2 < 2; ++kt2) {
            const unsigned short* pp = &lds_p[wave][m][kt2 * 32 + h * 8];
            ap[kt2] = frag_from(pp, pp + 16);
        }
#pragma unroll
        for (int f = 0; f < 8; ++f) {
            const unsigned short* vp0 = &lds_v[f * 16 + m][h * 8];
            const unsigned short* vp1 = &lds_v[f * 16 + m][32 + h * 8];
            v16bf bv0 = frag_from(vp0, vp0 + 16);
            v16bf bv1 = frag_from(vp1, vp1 + 16);
            o[f] = __builtin_amdgcn_wmma_f32_16x16x32_bf16(
                       false, ap[0], false, bv0, (short)0, o[f], false, false);
            o[f] = __builtin_amdgcn_wmma_f32_16x16x32_bf16(
                       false, ap[1], false, bv1, (short)0, o[f], false, false);
        }
    }

    // --- normalize and store (fp32 out, C layout scatter) ---
#pragma unroll
    for (int r = 0; r < 8; ++r) {
        const float inv = 1.0f / lrow[r];
        const int row = qbase + r + 8 * h;
        float* op = out + (size_t)(b * T_ + row) * DD;
#pragma unroll
        for (int f = 0; f < 8; ++f)
            op[f * 16 + m] = o[f][r] * inv;
    }
}

// ---------------------------------------------------------------------------
extern "C" void kernel_launch(void* const* d_in, const int* in_sizes, int n_in,
                              void* d_out, int out_size, void* d_ws, size_t ws_size,
                              hipStream_t stream) {
    const float* seq = (const float*)d_in[0];
    // d_in[1] = lengths (unused by reference computation)
    const float* Wq  = (const float*)d_in[2];
    const float* Wk  = (const float*)d_in[3];
    const float* Wv  = (const float*)d_in[4];

    // Workspace layout (bf16): qb/kb/vb 12 MB, seqb 8 MB, Wt 192 KB
    unsigned short* qb   = (unsigned short*)d_ws;
    unsigned short* kb   = qb + (size_t)B_ * T_ * DD;
    unsigned short* vb   = kb + (size_t)B_ * T_ * DD;
    unsigned short* seqb = vb + (size_t)B_ * T_ * DD;
    unsigned short* Wt   = seqb + (size_t)B_ * T_ * NIN;

    cvt_seq_kernel<<<(B_ * T_ * NIN / 4) / 256, 256, 0, stream>>>(seq, seqb);
    wt_transpose_kernel<<<(3 * NIN * DD) / 256, 256, 0, stream>>>(Wq, Wk, Wv, Wt);
    qkv_wmma_kernel<<<B_ * T_ / QROWS, 128, 0, stream>>>(seqb, Wt, qb, kb, vb);
    attn_kernel<<<B_ * (T_ / QROWS), WAVES * 32, 0, stream>>>(qb, kb, vb, (float*)d_out);
}
